// GCL_1056561954999
// MI455X (gfx1250) — matrix operand order repaired and checked by reading, not compiled, not run on hardware
//
#include <hip/hip_runtime.h>
#include <hip/hip_bf16.h>

#define HID 128
#define WAVES 4   // waves per block (128 threads)

typedef __attribute__((ext_vector_type(16))) _Float16 v16h;
typedef __attribute__((ext_vector_type(8)))  float    v8f;

__device__ __forceinline__ v8f zero8() {
  v8f z;
#pragma unroll
  for (int i = 0; i < 8; ++i) z[i] = 0.0f;
  return z;
}

__device__ __forceinline__ v8f wmma_f16(v16h a, v16h b, v8f c) {
  return __builtin_amdgcn_wmma_f32_16x16x32_f16(false, a, false, b, (short)0, c,
                                                false, false);
}

// Native fp32 atomic add (guaranteed global_atomic_add_f32, no CAS loop).
__device__ __forceinline__ void atomic_add_f32(float* p, float v) {
  asm volatile("global_atomic_add_f32 %0, %1, off" : : "v"(p), "v"(v) : "memory");
}

// Async global -> LDS copy (GLOBAL_LOAD_ASYNC_TO_LDS_B128, tracked by ASYNCcnt).
// Each lane moves 16 B per issue; whole block cooperates. Caller must
// async_wait() + __syncthreads() before reading the LDS region.
__device__ __forceinline__ void stage_async(char* dstLds, const char* srcGlb,
                                            int bytes, int tid, int nthreads) {
  unsigned l = (unsigned)(uintptr_t)dstLds + (unsigned)tid * 16u;  // low 32 = LDS offset
  const char* g = srcGlb + tid * 16;
  const int step = nthreads * 16;
  for (int off = tid * 16; off < bytes; off += step) {
    asm volatile("global_load_async_to_lds_b128 %0, %1, off"
                 : : "v"(l), "v"(g) : "memory");
    l += (unsigned)step;
    g += step;
  }
}
__device__ __forceinline__ void async_wait() {
  asm volatile("s_wait_asynccnt 0x0" ::: "memory");
}

// ---------------------------------------------------------------------------
// Weight packing: fp32 [K x 128] row-major -> f16 WMMA B-operand tiles.
// B 16-bit 32x16 layout: lane = 16*half + n holds K = 32*ks + 8*half + e (e<8), 32*ks + 16 + 8*half + (e-8) (e>=8).
// dst[(nt*KS + ks)*32 + lane][e] = W[that K][16*nt + n]
// ---------------------------------------------------------------------------
__global__ __launch_bounds__(32) void pack_w_kernel(const float* __restrict__ src,
                                                    v16h* __restrict__ dst, int KS) {
  const int lane = threadIdx.x;
  const int t    = blockIdx.x;       // t = nt*KS + ks
  const int ks   = t % KS;
  const int nt   = t / KS;
  const int half = lane >> 4;
  const int n    = lane & 15;
  v16h b;
#pragma unroll
  for (int e = 0; e < 16; ++e) {
    const int k = 32 * ks + (e < 8 ? 8 * half + e : 16 + 8 * half + (e - 8));
    b[e] = (_Float16)src[k * HID + nt * 16 + n];
  }
  dst[t * 32 + lane] = b;
}

__global__ __launch_bounds__(256) void zero_kernel(float* __restrict__ p, long long n) {
  long long i = (long long)blockIdx.x * blockDim.x + threadIdx.x;
  const long long stride = (long long)gridDim.x * blockDim.x;
  for (; i < n; i += stride) p[i] = 0.0f;
}

// ---------------------------------------------------------------------------
// Edge kernel: one wave = 16 edges.
//   diff = h[row]-h[col] -> relu(diff@W1+b1)@W2+b2 -> atomic scatter to agg[row]
// Dynamic LDS: [0,32K) W1 tiles, [32K,64K) W2 tiles, [64K,96K) transpose tiles.
// ---------------------------------------------------------------------------
__global__ __launch_bounds__(32 * WAVES) void edge_kernel(
    const float* __restrict__ h, const long long* __restrict__ ei,
    const float* __restrict__ b1v, const float* __restrict__ b2v,
    const v16h* __restrict__ pw1, const v16h* __restrict__ pw2,
    float* __restrict__ agg, int E, int nTiles) {
  extern __shared__ char smem[];
  v16h*  W1 = (v16h*)smem;                    // 1024 tiles-of-lane = 32 KB
  v16h*  W2 = (v16h*)(smem + 32768);
  float* tp = (float*)(smem + 65536);

  // --- block-cooperative async staging of weights into LDS ----------------
  stage_async(smem,         (const char*)pw1, 32768, threadIdx.x, blockDim.x);
  stage_async(smem + 32768, (const char*)pw2, 32768, threadIdx.x, blockDim.x);
  async_wait();
  __syncthreads();

  const int wid  = threadIdx.x >> 5;
  const int lane = threadIdx.x & 31;
  const int tile = blockIdx.x * WAVES + wid;
  if (tile >= nTiles) return;                 // wave-uniform
  const int half = lane >> 4;
  const int n    = lane & 15;
  const long long base = (long long)tile * 16;

  // --- gather this lane's A row (row m == n) and build f16 A fragments ----
  long long eA = base + n;
  if (eA >= E) eA = E - 1;
  const long long rA = ei[eA];
  const long long cA = ei[(long long)E + eA];
  const float* hr = h + rA * HID;
  const float* hc = h + cA * HID;

  v16h a[4];
#pragma unroll
  for (int ks = 0; ks < 4; ++ks) {
    const int off = 32 * ks + 8 * half;
#pragma unroll
    for (int i = 0; i < 8; ++i) a[ks][i]     = (_Float16)(hr[off + i]      - hc[off + i]);
#pragma unroll
    for (int i = 0; i < 8; ++i) a[ks][8 + i] = (_Float16)(hr[off + 16 + i] - hc[off + 16 + i]);
  }

  // --- GEMM1: [16x128] @ W1[128x128], B from LDS ---------------------------
  v8f acc[8];
#pragma unroll
  for (int nt = 0; nt < 8; ++nt) acc[nt] = zero8();
#pragma unroll
  for (int nt = 0; nt < 8; ++nt)
#pragma unroll
    for (int ks = 0; ks < 4; ++ks)
      acc[nt] = wmma_f16(a[ks], W1[(nt * 4 + ks) * 32 + lane], acc[nt]);

  // --- bias + ReLU, transpose C-layout -> row-major via per-wave LDS ------
  float* L = tp + wid * 16 * HID;
#pragma unroll
  for (int nt = 0; nt < 8; ++nt) {
    const float bv = b1v[nt * 16 + n];
#pragma unroll
    for (int v = 0; v < 8; ++v) {
      float x = acc[nt][v] + bv;
      L[(v + 8 * half) * HID + nt * 16 + n] = x > 0.0f ? x : 0.0f;
    }
  }
  asm volatile("s_wait_dscnt 0x0" ::: "memory");  // same-wave DS in-order

  // --- reload as A fragments (f16) ----------------------------------------
  v16h a2[4];
  {
    const float* Lr = L + n * HID;
#pragma unroll
    for (int ks = 0; ks < 4; ++ks) {
      const int off = 32 * ks + 8 * half;
#pragma unroll
      for (int i = 0; i < 8; ++i) a2[ks][i]     = (_Float16)Lr[off + i];
#pragma unroll
      for (int i = 0; i < 8; ++i) a2[ks][8 + i] = (_Float16)Lr[off + 16 + i];
    }
  }

  // --- GEMM2 ---------------------------------------------------------------
  v8f acc2[8];
#pragma unroll
  for (int nt = 0; nt < 8; ++nt) acc2[nt] = zero8();
#pragma unroll
  for (int nt = 0; nt < 8; ++nt)
#pragma unroll
    for (int ks = 0; ks < 4; ++ks)
      acc2[nt] = wmma_f16(a2[ks], W2[(nt * 4 + ks) * 32 + lane], acc2[nt]);

  // --- bias + scatter-add (segment_sum over row) ---------------------------
  long long rdst[8];
  bool ok[8];
#pragma unroll
  for (int v = 0; v < 8; ++v) {
    const long long e2 = base + v + 8 * half;
    ok[v]   = (e2 < E);
    rdst[v] = ei[ok[v] ? e2 : 0];
  }
#pragma unroll
  for (int nt = 0; nt < 8; ++nt) {
    const float bv = b2v[nt * 16 + n];
#pragma unroll
    for (int v = 0; v < 8; ++v)
      if (ok[v]) atomic_add_f32(agg + rdst[v] * HID + nt * 16 + n, acc2[nt][v] + bv);
  }
}

// ---------------------------------------------------------------------------
// Node kernel: one wave = 16 nodes.  z = [h | agg] (16x256) -> MLP -> out.
// Dynamic LDS: [0,64K) W1 tiles, [64K,96K) W2 tiles, [96K,128K) transpose.
// Safe even if agg aliases d_out (all loads precede all stores per wave).
// ---------------------------------------------------------------------------
__global__ __launch_bounds__(32 * WAVES) void node_kernel(
    const float* __restrict__ h, const float* __restrict__ agg,
    const float* __restrict__ b1v, const float* __restrict__ b2v,
    const v16h* __restrict__ pw1, const v16h* __restrict__ pw2,
    float* __restrict__ out, int N, int nTiles) {
  extern __shared__ char smem[];
  v16h*  W1 = (v16h*)smem;                    // 2048 tiles-of-lane = 64 KB
  v16h*  W2 = (v16h*)(smem + 65536);
  float* tp = (float*)(smem + 98304);

  stage_async(smem,         (const char*)pw1, 65536, threadIdx.x, blockDim.x);
  stage_async(smem + 65536, (const char*)pw2, 32768, threadIdx.x, blockDim.x);
  async_wait();
  __syncthreads();

  const int wid  = threadIdx.x >> 5;
  const int lane = threadIdx.x & 31;
  const int tile = blockIdx.x * WAVES + wid;
  if (tile >= nTiles) return;
  const int half = lane >> 4;
  const int n    = lane & 15;
  const long long base = (long long)tile * 16;

  long long node = base + n;
  if (node >= N) node = N - 1;
  const float* hr = h + node * HID;
  const float* ar = agg + node * HID;

  v16h a[8];
#pragma unroll
  for (int ks = 0; ks < 8; ++ks) {
    const float* src = (ks < 4) ? hr : ar;
    const int off = 32 * (ks & 3) + 8 * half;
#pragma unroll
    for (int i = 0; i < 8; ++i) a[ks][i]     = (_Float16)src[off + i];
#pragma unroll
    for (int i = 0; i < 8; ++i) a[ks][8 + i] = (_Float16)src[off + 16 + i];
  }

  v8f acc[8];
#pragma unroll
  for (int nt = 0; nt < 8; ++nt) acc[nt] = zero8();
#pragma unroll
  for (int nt = 0; nt < 8; ++nt)
#pragma unroll
    for (int ks = 0; ks < 8; ++ks)
      acc[nt] = wmma_f16(a[ks], W1[(nt * 8 + ks) * 32 + lane], acc[nt]);

  float* L = tp + wid * 16 * HID;
#pragma unroll
  for (int nt = 0; nt < 8; ++nt) {
    const float bv = b1v[nt * 16 + n];
#pragma unroll
    for (int v = 0; v < 8; ++v) {
      float x = acc[nt][v] + bv;
      L[(v + 8 * half) * HID + nt * 16 + n] = x > 0.0f ? x : 0.0f;
    }
  }
  asm volatile("s_wait_dscnt 0x0" ::: "memory");

  v16h a2[4];
  {
    const float* Lr = L + n * HID;
#pragma unroll
    for (int ks = 0; ks < 4; ++ks) {
      const int off = 32 * ks + 8 * half;
#pragma unroll
      for (int i = 0; i < 8; ++i) a2[ks][i]     = (_Float16)Lr[off + i];
#pragma unroll
      for (int i = 0; i < 8; ++i) a2[ks][8 + i] = (_Float16)Lr[off + 16 + i];
    }
  }

  v8f acc2[8];
#pragma unroll
  for (int nt = 0; nt < 8; ++nt) acc2[nt] = zero8();
#pragma unroll
  for (int nt = 0; nt < 8; ++nt)
#pragma unroll
    for (int ks = 0; ks < 4; ++ks)
      acc2[nt] = wmma_f16(a2[ks], W2[(nt * 4 + ks) * 32 + lane], acc2[nt]);

#pragma unroll
  for (int nt = 0; nt < 8; ++nt) {
    const float bv = b2v[nt * 16 + n];
#pragma unroll
    for (int v = 0; v < 8; ++v) {
      const long long nr = base + v + 8 * half;
      if (nr < N) out[nr * HID + nt * 16 + n] = acc2[nt][v] + bv;
    }
  }
}

// ---------------------------------------------------------------------------
extern "C" void kernel_launch(void* const* d_in, const int* in_sizes, int n_in,
                              void* d_out, int out_size, void* d_ws, size_t ws_size,
                              hipStream_t stream) {
  const float*     h   = (const float*)d_in[0];
  const long long* ei  = (const long long*)d_in[1];
  const float*     ew1 = (const float*)d_in[2];
  const float*     eb1 = (const float*)d_in[3];
  const float*     ew2 = (const float*)d_in[4];
  const float*     eb2 = (const float*)d_in[5];
  const float*     nw1 = (const float*)d_in[6];
  const float*     nb1 = (const float*)d_in[7];
  const float*     nw2 = (const float*)d_in[8];
  const float*     nb2 = (const float*)d_in[9];

  const int N = in_sizes[0] / HID;
  const int E = in_sizes[1] / 2;

  // Workspace layout: packed f16 weights (160 KB) then fp32 agg (N*128*4).
  char* ws  = (char*)d_ws;
  v16h* pe1 = (v16h*)(ws);                 // 128x128 -> 8*4*32 tiles = 32 KB
  v16h* pe2 = pe1 + 8 * 4 * 32;            // 32 KB
  v16h* pn1 = pe2 + 8 * 4 * 32;            // 256x128 -> 8*8*32 tiles = 64 KB
  v16h* pn2 = pn1 + 8 * 8 * 32;            // 32 KB
  const size_t wbytes = (size_t)(8 * 4 * 32 * 3 + 8 * 8 * 32) * sizeof(v16h); // 160 KB
  const size_t aggOff = (wbytes + 255) & ~(size_t)255;

  float* agg;
  const size_t need = aggOff + (size_t)N * HID * sizeof(float);
  if (ws_size >= need) agg = (float*)(ws + aggOff);
  else                 agg = (float*)d_out;   // fallback: accumulate in d_out

  pack_w_kernel<<<8 * 4, 32, 0, stream>>>(ew1, pe1, 4);
  pack_w_kernel<<<8 * 4, 32, 0, stream>>>(ew2, pe2, 4);
  pack_w_kernel<<<8 * 8, 32, 0, stream>>>(nw1, pn1, 8);
  pack_w_kernel<<<8 * 4, 32, 0, stream>>>(nw2, pn2, 4);

  zero_kernel<<<2048, 256, 0, stream>>>(agg, (long long)N * HID);

  const int eTiles = (E + 15) / 16;
  edge_kernel<<<(eTiles + WAVES - 1) / WAVES, 32 * WAVES, 98304, stream>>>(
      h, ei, eb1, eb2, pe1, pe2, agg, E, eTiles);

  const int nTiles = (N + 15) / 16;
  node_kernel<<<(nTiles + WAVES - 1) / WAVES, 32 * WAVES, 131072, stream>>>(
      h, agg, nb1, nb2, pn1, pn2, (float*)d_out, N, nTiles);
}
